// SpectraGNN_40450001994134
// MI455X (gfx1250) — matrix operand order repaired and checked by reading, not compiled, not run on hardware
//
#include <hip/hip_runtime.h>
#include <hip/hip_bf16.h>

typedef __bf16 bf16;
typedef __bf16 v16bf __attribute__((ext_vector_type(16)));
typedef float  v8f   __attribute__((ext_vector_type(8)));
typedef unsigned u32x4 __attribute__((ext_vector_type(4)));

#define NEG_SLOPE 0.2f
#define ENC_NEG_INF 0x007FFFFFu   // enc(-inf) for ordered-uint float max

// ---------------------------------------------------------------- helpers
__device__ __forceinline__ bf16 f2bf(float f) {
  unsigned u = __float_as_uint(f);
  unsigned r = (u + 0x7fffu + ((u >> 16) & 1u)) >> 16;   // RNE
  unsigned short s = (unsigned short)r;
  return __builtin_bit_cast(bf16, s);
}
__device__ __forceinline__ unsigned enc_ord(float v) {
  unsigned b = __float_as_uint(v);
  return (b & 0x80000000u) ? ~b : (b | 0x80000000u);
}
__device__ __forceinline__ float dec_ord(unsigned e) {
  return __uint_as_float((e & 0x80000000u) ? (e & 0x7fffffffu) : ~e);
}

// ---------------------------------------------------------------- kernels
__global__ void cvt_f32_bf16(const float* __restrict__ in, bf16* __restrict__ out,
                             long long n) {
  long long i = (long long)blockIdx.x * blockDim.x + threadIdx.x;
  if (i < n) out[i] = f2bf(in[i]);
}

__global__ void fill_u32k(unsigned* __restrict__ p, unsigned val, long long n) {
  long long i = (long long)blockIdx.x * blockDim.x + threadIdx.x;
  if (i < n) p[i] = val;
}

// C[M,Nout] = A[M,K](bf16) * W[Nout,K](bf16)^T, fp32 accumulate via WMMA.
// Block = 8 waves; wave (mw,nw) computes 16x16 tile; block tile 32(M) x 64(N).
__global__ void __launch_bounds__(256)
gat_gemm_bf16(const bf16* __restrict__ A, const bf16* __restrict__ W,
              float* __restrict__ C, int M, int K, int Nout)
{
  const int lane = threadIdx.x & 31;
  const int wave = threadIdx.x >> 5;
  const int m0 = blockIdx.x * 32 + (wave >> 2) * 16;
  const int n0 = blockIdx.y * 64 + (wave & 3) * 16;

  const int l16 = lane & 15;
  const int hi  = lane >> 4;          // which half-wave
  int row = m0 + l16; if (row >= M) row = M - 1;   // clamp (no divergence pre-WMMA)
  const int col = n0 + l16;

  // A 16x32 bf16 layout: lane<16 holds K {0..7,16..23}; lane>=16 holds {8..15,24..31}
  const bf16* ap = A + (size_t)row * K + hi * 8;
  // B 32x16 bf16 layout: lane<16 holds K {0..7,16..23}; lane>=16 holds {8..15,24..31}
  const bf16* bp = W + (size_t)col * K + hi * 8;

  v8f acc = {};
  for (int k0 = 0; k0 < K; k0 += 32) {
    union { v16bf v; u32x4 q[2]; } a, b;
    a.q[0] = *(const u32x4*)(ap + k0);
    a.q[1] = *(const u32x4*)(ap + k0 + 16);
    b.q[0] = *(const u32x4*)(bp + k0);
    b.q[1] = *(const u32x4*)(bp + k0 + 16);
    acc = __builtin_amdgcn_wmma_f32_16x16x32_bf16(false, a.v, false, b.v,
                                                  (short)0, acc, false, false);
  }
  // C/D layout: lane half selects M base (+8); VGPR v -> M = v (+8), N = lane%16
  #pragma unroll
  for (int v = 0; v < 8; ++v) {
    int m = m0 + v + hi * 8;
    if (m < M) C[(size_t)m * Nout + col] = acc[v];
  }
}

// per-(node,head) attention logits: as = <h, a_src>, ad = <h, a_dst>  (C = 64)
__global__ void gat_alpha(const float* __restrict__ h, const float* __restrict__ aw_s,
                          const float* __restrict__ aw_d, float* __restrict__ as,
                          float* __restrict__ ad, int Nn, int Hh)
{
  int i = blockIdx.x * blockDim.x + threadIdx.x;
  if (i >= Nn * Hh) return;
  int n = i / Hh, hh = i - n * Hh;
  const float* hp = h + ((size_t)n * Hh + hh) * 64;
  const float* sw = aw_s + hh * 64;
  const float* dw = aw_d + hh * 64;
  float vs = 0.f, vd = 0.f;
  #pragma unroll 8
  for (int c = 0; c < 64; ++c) { float x = hp[c]; vs += x * sw[c]; vd += x * dw[c]; }
  as[i] = vs; ad[i] = vd;
}

__global__ void gat_edge_max(const int* __restrict__ ei, int E, int Ep, int Hh,
                             const float* __restrict__ as, const float* __restrict__ ad,
                             unsigned* __restrict__ emax)
{
  int e = blockIdx.x * blockDim.x + threadIdx.x;
  if (e >= Ep) return;
  int s = (e < E) ? ei[e] : (e - E);
  int d = (e < E) ? ei[E + e] : (e - E);
  for (int h = 0; h < Hh; ++h) {
    float v = as[s * Hh + h] + ad[d * Hh + h];
    v = (v > 0.f) ? v : NEG_SLOPE * v;
    atomicMax(&emax[d * Hh + h], enc_ord(v));
  }
}

__global__ void gat_edge_exp(const int* __restrict__ ei, int E, int Ep, int Hh,
                             const float* __restrict__ as, const float* __restrict__ ad,
                             const unsigned* __restrict__ emax, float* __restrict__ denom,
                             float* __restrict__ exbuf)
{
  int e = blockIdx.x * blockDim.x + threadIdx.x;
  if (e >= Ep) return;
  int s = (e < E) ? ei[e] : (e - E);
  int d = (e < E) ? ei[E + e] : (e - E);
  for (int h = 0; h < Hh; ++h) {
    float v = as[s * Hh + h] + ad[d * Hh + h];
    v = (v > 0.f) ? v : NEG_SLOPE * v;
    float ex = expf(v - dec_ord(emax[d * Hh + h]));
    exbuf[(size_t)e * Hh + h] = ex;
    atomicAdd(&denom[d * Hh + h], ex);
  }
}

// one block per edge, one thread per output feature (F = Hh*64)
__global__ void gat_aggregate(const int* __restrict__ ei, int E, int Ep, int Hh,
                              const float* __restrict__ exbuf, const float* __restrict__ denom,
                              const float* __restrict__ h, float* __restrict__ out, int F)
{
  int e = blockIdx.x;
  if (e >= Ep) return;
  int s = (e < E) ? ei[e] : (e - E);
  int d = (e < E) ? ei[E + e] : (e - E);
  int f = threadIdx.x;
  int hh = f >> 6;
  float coeff = exbuf[(size_t)e * Hh + hh] / (denom[(size_t)d * Hh + hh] + 1e-16f);
  atomicAdd(&out[(size_t)d * F + f], coeff * h[(size_t)s * F + f]);
}

__global__ void gat_bias_act(float* __restrict__ out, const float* __restrict__ bias,
                             bf16* __restrict__ obf, long long n, int F, int relu)
{
  long long i = (long long)blockIdx.x * blockDim.x + threadIdx.x;
  if (i >= n) return;
  int f = (int)(i % F);
  float v = out[i] + bias[f];
  if (relu) v = v > 0.f ? v : 0.f;
  out[i] = v;
  if (obf) obf[i] = f2bf(v);
}

__global__ void gat_pool(const float* __restrict__ h, const int* __restrict__ seg,
                         float* __restrict__ pooled, float* __restrict__ cnt, int Nn)
{
  int i = blockIdx.x * blockDim.x + threadIdx.x;
  if (i >= Nn * 64) return;
  int n = i >> 6, c = i & 63;
  int g = seg[n];
  atomicAdd(&pooled[g * 64 + c], h[i]);
  if (c == 0) atomicAdd(&cnt[g], 1.0f);
}

__global__ void gat_out_head(const float* __restrict__ pooled, const float* __restrict__ cnt,
                             const float* __restrict__ Wout, const float* __restrict__ bout,
                             float* __restrict__ out, int G, int T)
{
  int i = blockIdx.x * blockDim.x + threadIdx.x;
  if (i >= G * T) return;
  int g = i / T, t = i - g * T;
  float inv = 1.0f / fmaxf(cnt[g], 1.0f);
  const float* pg = pooled + g * 64;
  const float* wt = Wout + t * 64;
  float acc = bout[t];
  #pragma unroll 8
  for (int c = 0; c < 64; ++c) acc += (pg[c] * inv) * wt[c];
  out[i] = acc;
}

// ---------------------------------------------------------------- launch
extern "C" void kernel_launch(void* const* d_in, const int* in_sizes, int n_in,
                              void* d_out, int out_size, void* d_ws, size_t ws_size,
                              hipStream_t stream) {
  const float* x     = (const float*)d_in[0];
  const int*   ei    = (const int*)  d_in[1];
  const int*   seg   = (const int*)  d_in[2];
  const float* W0    = (const float*)d_in[3];
  const float* as0   = (const float*)d_in[4];
  const float* ad0   = (const float*)d_in[5];
  const float* b0    = (const float*)d_in[6];
  const float* W1    = (const float*)d_in[7];
  const float* as1   = (const float*)d_in[8];
  const float* ad1   = (const float*)d_in[9];
  const float* b1    = (const float*)d_in[10];
  const float* W2    = (const float*)d_in[11];
  const float* as2   = (const float*)d_in[12];
  const float* ad2   = (const float*)d_in[13];
  const float* b2    = (const float*)d_in[14];
  const float* Wout  = (const float*)d_in[15];
  const float* bout  = (const float*)d_in[16];

  const int FIN = 64, HC = 256;
  const int Nn = in_sizes[0] / FIN;
  const int E  = in_sizes[1] / 2;
  const int Ep = E + Nn;
  const int T  = in_sizes[16];
  const int G  = out_size / T;

  // workspace carve-up (256-B aligned)
  char* ws = (char*)d_ws;
  size_t off = 0;
  auto carve = [&](size_t bytes) { void* p = ws + off; off += (bytes + 255) & ~(size_t)255; return p; };
  float*    bufA  = (float*)   carve((size_t)Nn * HC * 4);   // GEMM output h
  float*    bufB  = (float*)   carve((size_t)Nn * HC * 4);   // aggregated out
  bf16*     hbf   = (bf16*)    carve((size_t)Nn * HC * 2);   // next-layer bf16 input
  bf16*     xbf   = (bf16*)    carve((size_t)Nn * FIN * 2);
  bf16*     w0b   = (bf16*)    carve((size_t)HC * FIN * 2);
  bf16*     w1b   = (bf16*)    carve((size_t)HC * HC * 2);
  bf16*     w2b   = (bf16*)    carve((size_t)64 * HC * 2);
  float*    asb   = (float*)   carve((size_t)Nn * 4 * 4);
  float*    adb   = (float*)   carve((size_t)Nn * 4 * 4);
  unsigned* emax  = (unsigned*)carve((size_t)Nn * 4 * 4);
  float*    denom = (float*)   carve((size_t)Nn * 4 * 4);
  float*    exbuf = (float*)   carve((size_t)Ep * 4 * 4);
  float*    pooled= (float*)   carve((size_t)G * 64 * 4);
  float*    cnt   = (float*)   carve((size_t)G * 4);

  auto g1 = [](long long n) { return dim3((unsigned)((n + 255) / 256)); };

  // bf16 copies of weights and input features
  cvt_f32_bf16<<<g1((long long)Nn * FIN), 256, 0, stream>>>(x,  xbf, (long long)Nn * FIN);
  cvt_f32_bf16<<<g1((long long)HC * FIN), 256, 0, stream>>>(W0, w0b, (long long)HC * FIN);
  cvt_f32_bf16<<<g1((long long)HC * HC),  256, 0, stream>>>(W1, w1b, (long long)HC * HC);
  cvt_f32_bf16<<<g1((long long)64 * HC),  256, 0, stream>>>(W2, w2b, (long long)64 * HC);

  auto run_layer = [&](const bf16* inbf, int K, const bf16* Wb, int Fout, int Hh,
                       const float* aws, const float* awd, const float* bias,
                       bf16* nextbf, int relu) {
    dim3 gg((Nn + 31) / 32, Fout / 64);
    gat_gemm_bf16<<<gg, 256, 0, stream>>>(inbf, Wb, bufA, Nn, K, Fout);

    long long nh = (long long)Nn * Hh;
    gat_alpha<<<g1(nh), 256, 0, stream>>>(bufA, aws, awd, asb, adb, Nn, Hh);

    fill_u32k<<<g1(nh), 256, 0, stream>>>(emax, ENC_NEG_INF, nh);
    fill_u32k<<<g1(nh), 256, 0, stream>>>((unsigned*)denom, 0u, nh);
    fill_u32k<<<g1((long long)Nn * Fout), 256, 0, stream>>>((unsigned*)bufB, 0u,
                                                            (long long)Nn * Fout);

    gat_edge_max<<<g1(Ep), 256, 0, stream>>>(ei, E, Ep, Hh, asb, adb, emax);
    gat_edge_exp<<<g1(Ep), 256, 0, stream>>>(ei, E, Ep, Hh, asb, adb, emax, denom, exbuf);
    gat_aggregate<<<dim3((unsigned)Ep), dim3((unsigned)Fout), 0, stream>>>(
        ei, E, Ep, Hh, exbuf, denom, bufA, bufB, Fout);

    gat_bias_act<<<g1((long long)Nn * Fout), 256, 0, stream>>>(
        bufB, bias, nextbf, (long long)Nn * Fout, Fout, relu);
  };

  run_layer(xbf, FIN, w0b, HC, 4, as0, ad0, b0, hbf, 1);   // GAT(64 -> 64, H=4) + relu
  run_layer(hbf, HC,  w1b, HC, 4, as1, ad1, b1, hbf, 1);   // GAT(256 -> 64, H=4) + relu
  run_layer(hbf, HC,  w2b, 64, 1, as2, ad2, b2, nullptr, 0); // GAT(256 -> 64, H=1)

  // global mean pool + output head
  fill_u32k<<<g1((long long)G * 64), 256, 0, stream>>>((unsigned*)pooled, 0u, (long long)G * 64);
  fill_u32k<<<g1(G), 256, 0, stream>>>((unsigned*)cnt, 0u, G);
  gat_pool<<<g1((long long)Nn * 64), 256, 0, stream>>>(bufB, seg, pooled, cnt, Nn);
  gat_out_head<<<g1((long long)G * T), 256, 0, stream>>>(pooled, cnt, Wout, bout,
                                                         (float*)d_out, G, T);
}
